// GAU_20753281974688
// MI455X (gfx1250) — compile-verified, matrix operand order checked by reading, not run on hardware
//
#include <hip/hip_runtime.h>

// ---------------- problem constants (from reference) ----------------
#define B_   8
#define D_   192
#define H_   128
#define W_   128
#define S_   64          // codebook size
#define ST   8           // stride
#define RB   256         // blocks per batch = (H/ST)*(W/ST)
#define TOK  64          // tokens per block = ST*ST
#define DV   384         // 2*D
#define HWW  (H_*W_)
#define NBLK (B_*RB)
#define L_   (NBLK*TOK)
#define SCALE_QK 0.8408964152537145f   // 2^-0.25
#define EPS_ 1e-6f

typedef _Float16 h16;
typedef __attribute__((ext_vector_type(16))) _Float16 v16h;
typedef __attribute__((ext_vector_type(8)))  _Float16 v8h;
typedef __attribute__((ext_vector_type(8)))  float    v8f;

// ---- WMMA fragment load: 16-bit A/B operand, 16x32 tile from a row-major
// [rows][ld] f16 buffer (LDS or global). ISA layout (CDNA5 ISA §7.12.2):
// lane<16: row=lane, K pattern {k0..k0+7} in VGPR0-3, {k0+16..k0+23} in VGPR4-7
// lane>=16: row=lane-16, K pattern shifted by +8.
__device__ __forceinline__ v16h frag_ld(const h16* p, int ld, int r0, int k0) {
  const int lane = threadIdx.x & 31;
  const h16* q = p + (size_t)(r0 + (lane & 15)) * ld + (k0 + ((lane >> 4) << 3));
  v8h lo = *(const v8h*)(q);        // 8 contiguous halves (16B)
  v8h hi = *(const v8h*)(q + 16);   // next K group (+16)
  v16h f;
#pragma unroll
  for (int i = 0; i < 8; ++i) { f[i] = lo[i]; f[i + 8] = hi[i]; }
  return f;
}

__device__ __forceinline__ v8f wmma_f16(v16h a, v16h b, v8f c) {
  return __builtin_amdgcn_wmma_f32_16x16x32_f16(false, a, false, b,
                                                (short)0, c, false, false);
}

// fast reciprocal: single v_rcp_f32 instead of IEEE div expansion
__device__ __forceinline__ float fast_rcp(float v) { return __builtin_amdgcn_rcpf(v); }
__device__ __forceinline__ float silu_f(float v) { return v * fast_rcp(1.f + __expf(-v)); }

// ---------------- kernel 0: weight convert/transpose + zero bins ----------------
__global__ void GAU_prep(const float* __restrict__ Wz1, const float* __restrict__ Wv,
                         const float* __restrict__ Wg,  const float* __restrict__ Wc,
                         h16* wz1t, h16* wvt, h16* wgt, h16* wct,
                         float* dtv, float* dt1) {
  int i = blockIdx.x * 256 + threadIdx.x;
  if (i < D_ * D_) {                 // Wz1^T: [n=192][k=192]
    int n = i / D_, k = i % D_;
    wz1t[i] = (h16)Wz1[k * D_ + n];  return;
  }
  i -= D_ * D_;
  if (i < DV * D_) {                 // Wv^T: [n=384][k=192]
    int n = i / D_, k = i % D_;
    wvt[i] = (h16)Wv[k * DV + n];    return;
  }
  i -= DV * D_;
  if (i < DV * D_) {                 // Wg^T
    int n = i / D_, k = i % D_;
    wgt[i] = (h16)Wg[k * DV + n];    return;
  }
  i -= DV * D_;
  if (i < D_ * DV) {                 // Wc^T: [n=192][k=384]
    int n = i / DV, k = i % DV;
    wct[i] = (h16)Wc[k * D_ + n];    return;
  }
  i -= D_ * DV;
  if (i < B_ * S_ * DV) { dtv[i] = 0.f; return; }
  i -= B_ * S_ * DV;
  if (i < B_ * S_) dt1[i] = 0.f;
}

// ---------------- kernel 1: InstanceNorm stats ----------------
__global__ void GAU_stats(const float* __restrict__ x, float* mu, float* rstd) {
  const int bc = blockIdx.x;                     // b*D + c
  const float* p = x + (size_t)bc * HWW;
  float s = 0.f, ss = 0.f;
  for (int i = threadIdx.x; i < HWW; i += 256) { float v = p[i]; s += v; ss += v * v; }
  __shared__ float rs[256], rq[256];
  rs[threadIdx.x] = s; rq[threadIdx.x] = ss; __syncthreads();
  for (int o = 128; o > 0; o >>= 1) {
    if (threadIdx.x < o) { rs[threadIdx.x] += rs[threadIdx.x + o]; rq[threadIdx.x] += rq[threadIdx.x + o]; }
    __syncthreads();
  }
  if (threadIdx.x == 0) {
    float m = rs[0] * (1.f / HWW);
    float v = rq[0] * (1.f / HWW) - m * m;
    mu[bc] = m; rstd[bc] = rsqrtf(v + EPS_);
  }
}

// ---------------- kernel 2 (front pass): z -> q,k -> delta; scatter bins ----------------
// grid = NBLK, block = 256 (8 wave32), dyn LDS = 75008 B
__global__ void GAU_front(const float* __restrict__ x,
                          const float* __restrict__ mu, const float* __restrict__ rstd,
                          const h16* __restrict__ wz1t, const h16* __restrict__ wvt,
                          const float* __restrict__ Wz2,
                          const float* __restrict__ gq, const float* __restrict__ bq,
                          const float* __restrict__ gk, const float* __restrict__ bk,
                          const float* __restrict__ cb,
                          float* __restrict__ wsq, float* __restrict__ wsk,
                          int* __restrict__ wsd,
                          float* __restrict__ dtv, float* __restrict__ dt1) {
  extern __shared__ char smem[];
  h16*   xn   = (h16*)(smem);                 // [64][192]  24576B
  float* hbuf = (float*)(smem + 24576);       // [64][192]  49152B
  float* kl   = (float*)(smem + 73728);       // [64][2]
  float* cbl  = (float*)(smem + 74240);       // [64][2]
  int*   dl   = (int*)(smem + 74752);         // [64]

  const int tid = threadIdx.x, wave = tid >> 5;
  const int blk = blockIdx.x, b = blk / RB, r = blk % RB;
  const int bh = r / (W_ / ST), bw = r % (W_ / ST);

  // normalize x block -> xn (f16)
  for (int i = tid; i < TOK * D_; i += 256) {
    int t = i / D_, d = i % D_;
    int hh = bh * ST + (t >> 3), ww = bw * ST + (t & 7);
    float xv = x[(((size_t)b * D_ + d) * H_ + hh) * W_ + ww];
    xn[t * D_ + d] = (h16)((xv - mu[b * D_ + d]) * rstd[b * D_ + d]);
  }
  if (tid < 2 * S_) cbl[tid] = cb[tid];
  __syncthreads();

  // h = silu(xn @ Wz1): M=64 N=192 K=192
  for (int tile = wave; tile < 4 * 12; tile += 8) {
    int m0 = (tile & 3) * 16, n0 = (tile >> 2) * 16;
    v8f acc = {0, 0, 0, 0, 0, 0, 0, 0};
#pragma unroll
    for (int kk = 0; kk < D_; kk += 32)
      acc = wmma_f16(frag_ld(xn, D_, m0, kk), frag_ld(wz1t, D_, n0, kk), acc);
    int lane = tid & 31, n = n0 + (lane & 15), mb = m0 + ((lane >> 4) << 3);
#pragma unroll
    for (int rr = 0; rr < 8; ++rr) hbuf[(mb + rr) * D_ + n] = silu_f(acc[rr]);
  }
  __syncthreads();

  // z, q, k  (128 threads: token t, component j)
  if (tid < 128) {
    int t = tid >> 1, j = tid & 1;
    float z = 0.f;
    for (int d = 0; d < D_; ++d) z += hbuf[t * D_ + d] * Wz2[d * 2 + j];
    float qv = (z * gq[j] + bq[j]) * SCALE_QK;
    float kv = (z * gk[j] + bk[j]) * SCALE_QK;
    wsq[((size_t)blk * TOK + t) * 2 + j] = qv;
    wsk[((size_t)blk * TOK + t) * 2 + j] = kv;
    kl[t * 2 + j] = kv;
  }
  __syncthreads();

  // VQ argmin + count scatter
  if (tid < TOK) {
    float k0 = kl[tid * 2], k1 = kl[tid * 2 + 1];
    int best = 0; float bd = 3.4e38f;
    for (int s = 0; s < S_; ++s) {
      float d0 = k0 - cbl[s * 2], d1 = k1 - cbl[s * 2 + 1];
      float dd = d0 * d0 + d1 * d1;
      if (dd < bd) { bd = dd; best = s; }
    }
    dl[tid] = best;
    wsd[(size_t)blk * TOK + tid] = best;
    atomicAdd(&dt1[b * S_ + best], 1.0f);
  }
  __syncthreads();

  // v = silu(xn @ Wv): M=64 N=384 K=192 ; scatter rows into deltaTv bins
  for (int tile = wave; tile < 4 * 24; tile += 8) {
    int m0 = (tile & 3) * 16, n0 = (tile >> 2) * 16;
    v8f acc = {0, 0, 0, 0, 0, 0, 0, 0};
#pragma unroll
    for (int kk = 0; kk < D_; kk += 32)
      acc = wmma_f16(frag_ld(xn, D_, m0, kk), frag_ld(wvt, D_, n0, kk), acc);
    int lane = tid & 31, n = n0 + (lane & 15), mb = m0 + ((lane >> 4) << 3);
#pragma unroll
    for (int rr = 0; rr < 8; ++rr) {
      int m = mb + rr;
      atomicAdd(&dtv[((size_t)b * S_ + dl[m]) * DV + n], silu_f(acc[rr]));
    }
  }
}

// ---------------- kernel 3: transpose bins to f16 [b][n=384][s=64] ----------------
__global__ void GAU_binsT(const float* __restrict__ dtv, h16* __restrict__ dtvT) {
  int i = blockIdx.x * 256 + threadIdx.x;
  if (i >= B_ * DV * S_) return;
  int b = i / (DV * S_), rem = i % (DV * S_), n = rem / S_, s = rem % S_;
  dtvT[i] = (h16)dtv[((size_t)b * S_ + s) * DV + n];
}

// ---------------- kernel 4 (back pass): attention + output GEMM + residual --------
// grid = NBLK, block = 256, dyn LDS = 190976 B (WGP has 320 KB)
__global__ void GAU_back(const float* __restrict__ x,
                         const float* __restrict__ mu, const float* __restrict__ rstd,
                         const h16* __restrict__ wvt, const h16* __restrict__ wgt,
                         const h16* __restrict__ wct,
                         const float* __restrict__ wsq, const float* __restrict__ wsk,
                         const int* __restrict__ wsd,
                         const h16* __restrict__ dtvT, const float* __restrict__ dt1,
                         const float* __restrict__ cb,
                         float* __restrict__ out) {
  extern __shared__ char smem[];
  h16*   xn   = (h16*)(smem);                 // [64][192]   24576
  h16*   vT   = (h16*)(smem + 24576);         // [384][64]   49152  (v transposed: B operand)
  h16*   gb   = (h16*)(smem + 73728);         // [64][384]   49152
  h16*   qexp = (h16*)(smem + 122880);        // [64][64]    8192   (A operand)
  h16*   dif  = (h16*)(smem + 131072);        // [64][64]    8192   (A operand)
  h16*   awc  = (h16*)(smem + 139264);        // [64][384]   49152  (A operand)
  float* ql   = (float*)(smem + 188416);      // [64][2]
  float* kl   = (float*)(smem + 188928);      // [64][2]
  float* khl  = (float*)(smem + 189440);      // [64][2]
  float* rden = (float*)(smem + 189952);      // [64]  (stores 1/den)
  float* cbl  = (float*)(smem + 190208);      // [64][2]
  int*   dl   = (int*)(smem + 190720);        // [64]

  const int tid = threadIdx.x, wave = tid >> 5;
  const int blk = blockIdx.x, b = blk / RB, r = blk % RB;
  const int bh = r / (W_ / ST), bw = r % (W_ / ST);

  // loads: xn (normalize again), codebook, q/k/delta
  for (int i = tid; i < TOK * D_; i += 256) {
    int t = i / D_, d = i % D_;
    int hh = bh * ST + (t >> 3), ww = bw * ST + (t & 7);
    float xv = x[(((size_t)b * D_ + d) * H_ + hh) * W_ + ww];
    xn[t * D_ + d] = (h16)((xv - mu[b * D_ + d]) * rstd[b * D_ + d]);
  }
  if (tid < 2 * S_) cbl[tid] = cb[tid];
  if (tid < 128) {
    ql[tid] = wsq[(size_t)blk * TOK * 2 + tid];
    kl[tid] = wsk[(size_t)blk * TOK * 2 + tid];
  }
  if (tid < TOK) dl[tid] = wsd[(size_t)blk * TOK + tid];
  __syncthreads();

  if (tid < TOK) { khl[tid * 2] = cbl[dl[tid] * 2]; khl[tid * 2 + 1] = cbl[dl[tid] * 2 + 1]; }
  __syncthreads();

  // qexp[c][s], diff[c][j], 1/den[c]  (one token per thread)
  if (tid < TOK) {
    float q0 = ql[tid * 2], q1 = ql[tid * 2 + 1];
    float dn = 0.f;
    const float* cnt = dt1 + b * S_;
    for (int s = 0; s < S_; ++s) {
      float e = __expf(q0 * cbl[s * 2] + q1 * cbl[s * 2 + 1]);
      qexp[tid * S_ + s] = (h16)e;
      dn += e * cnt[s];
    }
    for (int j = 0; j < TOK; ++j) {
      float ae = __expf(q0 * kl[j * 2]  + q1 * kl[j * 2 + 1]);
      float ah = __expf(q0 * khl[j * 2] + q1 * khl[j * 2 + 1]);
      float df = ae - ah;
      dif[tid * TOK + j] = (h16)df;
      dn += df;
    }
    rden[tid] = fast_rcp(dn);
  }
  __syncthreads();

  // v = silu(xn@Wv) -> vT (transposed), g = silu(xn@Wg) -> gb
  for (int tile = wave; tile < 4 * 24; tile += 8) {
    int m0 = (tile & 3) * 16, n0 = (tile >> 2) * 16;
    v8f av = {0, 0, 0, 0, 0, 0, 0, 0}, ag = {0, 0, 0, 0, 0, 0, 0, 0};
#pragma unroll
    for (int kk = 0; kk < D_; kk += 32) {
      v16h a = frag_ld(xn, D_, m0, kk);
      av = wmma_f16(a, frag_ld(wvt, D_, n0, kk), av);
      ag = wmma_f16(a, frag_ld(wgt, D_, n0, kk), ag);
    }
    int lane = tid & 31, n = n0 + (lane & 15), mb = m0 + ((lane >> 4) << 3);
#pragma unroll
    for (int rr = 0; rr < 8; ++rr) {
      int m = mb + rr;
      vT[n * TOK + m]  = (h16)silu_f(av[rr]);
      gb[m * DV + n]   = (h16)silu_f(ag[rr]);
    }
  }
  __syncthreads();

  // num = qexp @ deltaTv^T  +  diff @ v   (M=64 N=384 K=64 each); wv*g -> awc
  const h16* dtvTb = dtvT + (size_t)b * DV * S_;
  for (int tile = wave; tile < 4 * 24; tile += 8) {
    int m0 = (tile & 3) * 16, n0 = (tile >> 2) * 16;
    v8f acc = {0, 0, 0, 0, 0, 0, 0, 0};
#pragma unroll
    for (int kk = 0; kk < S_; kk += 32)
      acc = wmma_f16(frag_ld(qexp, S_, m0, kk), frag_ld(dtvTb, S_, n0, kk), acc);
#pragma unroll
    for (int kk = 0; kk < TOK; kk += 32)
      acc = wmma_f16(frag_ld(dif, TOK, m0, kk), frag_ld(vT, TOK, n0, kk), acc);
    int lane = tid & 31, n = n0 + (lane & 15), mb = m0 + ((lane >> 4) << 3);
#pragma unroll
    for (int rr = 0; rr < 8; ++rr) {
      int m = mb + rr;
      float wv = acc[rr] * rden[m];
      awc[m * DV + n] = (h16)(wv * (float)gb[m * DV + n]);
    }
  }
  __syncthreads();

  // o = awc @ Wc : M=64 N=192 K=384 ; write featuremap + residual
  for (int tile = wave; tile < 4 * 12; tile += 8) {
    int m0 = (tile & 3) * 16, n0 = (tile >> 2) * 16;
    v8f acc = {0, 0, 0, 0, 0, 0, 0, 0};
#pragma unroll
    for (int kk = 0; kk < DV; kk += 32)
      acc = wmma_f16(frag_ld(awc, DV, m0, kk), frag_ld(wct, DV, n0, kk), acc);
    int lane = tid & 31, n = n0 + (lane & 15), mb = m0 + ((lane >> 4) << 3);
#pragma unroll
    for (int rr = 0; rr < 8; ++rr) {
      int m = mb + rr;
      int hh = bh * ST + (m >> 3), ww = bw * ST + (m & 7);
      size_t idx = (((size_t)b * D_ + n) * H_ + hh) * W_ + ww;
      out[idx] = acc[rr] + x[idx];
    }
  }
}

// ---------------- host side ----------------
extern "C" void kernel_launch(void* const* d_in, const int* in_sizes, int n_in,
                              void* d_out, int out_size, void* d_ws, size_t ws_size,
                              hipStream_t stream) {
  const float* x   = (const float*)d_in[0];
  const float* Wz1 = (const float*)d_in[1];
  const float* Wz2 = (const float*)d_in[2];
  const float* gq  = (const float*)d_in[3];
  const float* bq  = (const float*)d_in[4];
  const float* gk  = (const float*)d_in[5];
  const float* bk  = (const float*)d_in[6];
  const float* Wv  = (const float*)d_in[7];
  const float* Wg  = (const float*)d_in[8];
  const float* Wc  = (const float*)d_in[9];
  const float* cb  = (const float*)d_in[10];
  float* out = (float*)d_out;

  char* ws = (char*)d_ws;
  float* mu   = (float*)(ws + 0);                // 1536 f
  float* rstd = (float*)(ws + 6144);             // 1536 f
  float* wsq  = (float*)(ws + 12288);            // L*2 f
  float* wsk  = (float*)(ws + 1060864);          // L*2 f
  int*   wsd  = (int*)  (ws + 2109440);          // L   i
  float* dtv  = (float*)(ws + 2633728);          // B*S*DV f
  float* dt1  = (float*)(ws + 3420160);          // B*S f
  h16*   wz1t = (h16*)  (ws + 3422208);          // 192*192 h
  h16*   wvt  = (h16*)  (ws + 3495936);          // 384*192 h
  h16*   wgt  = (h16*)  (ws + 3643392);          // 384*192 h
  h16*   wct  = (h16*)  (ws + 3790848);          // 192*384 h
  h16*   dtvT = (h16*)  (ws + 3938304);          // B*384*64 h

  const int prepN = D_*D_ + 3*DV*D_ + B_*S_*DV + B_*S_;
  GAU_prep <<<(prepN + 255) / 256, 256, 0, stream>>>(Wz1, Wv, Wg, Wc,
                                                     wz1t, wvt, wgt, wct, dtv, dt1);
  GAU_stats<<<B_ * D_, 256, 0, stream>>>(x, mu, rstd);

  const size_t smemA = 75008;
  GAU_front<<<NBLK, 256, smemA, stream>>>(x, mu, rstd, wz1t, wvt, Wz2,
                                          gq, bq, gk, bk, cb,
                                          wsq, wsk, wsd, dtv, dt1);

  GAU_binsT<<<(B_ * DV * S_ + 255) / 256, 256, 0, stream>>>(dtv, dtvT);

  const size_t smemB = 190976;
  GAU_back <<<NBLK, 256, smemB, stream>>>(x, mu, rstd, wvt, wgt, wct,
                                          wsq, wsk, wsd, dtvT, dt1, cb, out);
}
